// SASRec_81750407512132
// MI455X (gfx1250) — matrix operand order repaired and checked by reading, not compiled, not run on hardware
//
#include <hip/hip_runtime.h>
#include <hip/hip_bf16.h>
#include <math.h>

// ---------------- problem constants ----------------
#define B_    128
#define T_    200
#define BT_   (B_ * T_)          // 25600 tokens, multiple of 64
#define H_    128
#define DH_   64
#define NH_   2
#define TXT_D 768
#define IMG_D 512
#define TT_   13                 // ceil(T_/16)
#define L_    2

typedef __attribute__((ext_vector_type(16))) _Float16 v16h;
typedef __attribute__((ext_vector_type(8)))  _Float16 v8h;
typedef __attribute__((ext_vector_type(8)))  float    v8f;

// ---------------- WMMA helper ----------------
__device__ __forceinline__ v8f wmma_f16(v16h a, v16h b, v8f c) {
  // D = A(16x32 f16) * B(32x16 f16) + C(16x16 f32)
  return __builtin_amdgcn_wmma_f32_16x16x32_f16(false, a, false, b, (short)0, c, false, false);
}

// A fragment (16x32 f16) from global fp32 row-major A[M x lda].
// Layout = two contiguous 8-element K-runs per lane: [k0+kb, +8) and
// [k0+kb+8, +8), kb=(l>>4)*16.  Runs are 8-aligned; Kmax must be a multiple
// of 8 so runs never straddle the bound.  lda*4 and run starts 16B-aligned.
__device__ __forceinline__ void load_a_glb(const float* A, int lda, int m0, int k0,
                                           int Kmax, int Mmax, v16h& a) {
  const int l  = threadIdx.x & 31;
  const int m  = m0 + (l & 15);
  const int kb = (l >> 4) * 16;
  const bool mv = (m < Mmax);
  const float* row = A + (size_t)m * lda;
  const int kA = k0 + kb, kB = k0 + kb + 8;
  float4 p0 = {0,0,0,0}, p1 = {0,0,0,0}, p2 = {0,0,0,0}, p3 = {0,0,0,0};
  if (mv && kA + 8 <= Kmax) { p0 = *(const float4*)(row + kA); p1 = *(const float4*)(row + kA + 4); }
  if (mv && kB + 8 <= Kmax) { p2 = *(const float4*)(row + kB); p3 = *(const float4*)(row + kB + 4); }
  a[0]=(_Float16)p0.x;  a[1]=(_Float16)p0.y;  a[2]=(_Float16)p0.z;  a[3]=(_Float16)p0.w;
  a[4]=(_Float16)p1.x;  a[5]=(_Float16)p1.y;  a[6]=(_Float16)p1.z;  a[7]=(_Float16)p1.w;
  a[8]=(_Float16)p2.x;  a[9]=(_Float16)p2.y;  a[10]=(_Float16)p2.z; a[11]=(_Float16)p2.w;
  a[12]=(_Float16)p3.x; a[13]=(_Float16)p3.y; a[14]=(_Float16)p3.z; a[15]=(_Float16)p3.w;
}

// A fragment from f16 (LDS or global), A[16 x lda]: two v8h runs per lane.
__device__ __forceinline__ void load_a_f16(const _Float16* A, int lda, int k0, v16h& a) {
  const int l  = threadIdx.x & 31;
  const int m  = l & 15;
  const int kb = (l >> 4) * 16;
  const _Float16* base = A + m * lda + k0 + kb;
  v8h lo = *(const v8h*)(base);
  v8h hi = *(const v8h*)(base + 8);
#pragma unroll
  for (int j = 0; j < 8; ++j) { a[j] = lo[j]; a[8 + j] = hi[j]; }
}

// B fragment (32x16 f16) from pre-transposed f16 weights WT[N x K] (row n = col n).
// Per lane: 16 contiguous halves at WT[n*K + k0 + kb], kb=(l>>4)*16 -> one v16h.
__device__ __forceinline__ void load_b_wt(const _Float16* WT, int K, int k0, int n0, v16h& b) {
  const int l  = threadIdx.x & 31;
  const int n  = n0 + (l & 15);
  const int kb = (l >> 4) * 16;
  b = *(const v16h*)(WT + (size_t)n * K + k0 + kb);
}

// B fragment from global fp32 row-major W[K x ldb] (pairs stride ldb apart; used
// only for the attention V matrix, coalesced across lanes).
__device__ __forceinline__ void load_b_glb(const float* W, int ldb, int k0, int n0,
                                           int Kmax, int Nmax, v16h& b) {
  const int l  = threadIdx.x & 31;
  const int n  = n0 + (l & 15);
  const int kb = (l >> 4) * 16;
  const bool nv = (n < Nmax);
#pragma unroll
  for (int j = 0; j < 8; ++j) {
    int k = k0 + kb + 2 * j;
    float x0 = (nv && k     < Kmax) ? W[(size_t)k * ldb + n]       : 0.f;
    float x1 = (nv && (k+1) < Kmax) ? W[(size_t)(k + 1) * ldb + n] : 0.f;
    b[2 * j]     = (_Float16)x0;
    b[2 * j + 1] = (_Float16)x1;
  }
}

// B fragment where logical B[k][n] = P[n*ld + k] (q @ k^T): per lane the 16
// k-values are one contiguous 16-float run -> 4x float4. Kmax=64 covers all runs.
__device__ __forceinline__ void load_bt_glb(const float* P, int ld, int k0, int n0,
                                            int Nmax, v16h& b) {
  const int l  = threadIdx.x & 31;
  const int n  = n0 + (l & 15);
  const int kb = (l >> 4) * 16;
  const bool nv = (n < Nmax);
  const float* base = P + (size_t)n * ld + k0 + kb;
  float4 p0 = {0,0,0,0}, p1 = {0,0,0,0}, p2 = {0,0,0,0}, p3 = {0,0,0,0};
  if (nv) {
    p0 = *(const float4*)(base);
    p1 = *(const float4*)(base + 4);
    p2 = *(const float4*)(base + 8);
    p3 = *(const float4*)(base + 12);
  }
  b[0]=(_Float16)p0.x;  b[1]=(_Float16)p0.y;  b[2]=(_Float16)p0.z;  b[3]=(_Float16)p0.w;
  b[4]=(_Float16)p1.x;  b[5]=(_Float16)p1.y;  b[6]=(_Float16)p1.z;  b[7]=(_Float16)p1.w;
  b[8]=(_Float16)p2.x;  b[9]=(_Float16)p2.y;  b[10]=(_Float16)p2.z; b[11]=(_Float16)p2.w;
  b[12]=(_Float16)p3.x; b[13]=(_Float16)p3.y; b[14]=(_Float16)p3.z; b[15]=(_Float16)p3.w;
}

// ---------------- weight transpose+convert: WT[n*K + k] = (f16) W[k*N + n] ----------
__global__ void wcvtT_kernel(const float* __restrict__ W, _Float16* __restrict__ WT,
                             int K, int N) {
  int idx = blockIdx.x * 256 + threadIdx.x;
  if (idx >= K * N) return;
  int k = idx / N, n = idx - k * N;
  WT[(size_t)n * K + k] = (_Float16)W[idx];
}

// ---------------- item embedding tower ----------------
// 16 tokens per block, 128 threads (4 waves). out[tok*H_ + h] fp32.
__global__ void __launch_bounds__(128)
item_emb_kernel(const int* __restrict__ ids_all,
                const float* __restrict__ txt_table, const float* __restrict__ img_table,
                const float* __restrict__ txt_ln_s, const float* __restrict__ txt_ln_b,
                const _Float16* __restrict__ txt_wT, const float* __restrict__ txt_b,
                const float* __restrict__ img_ln_s, const float* __restrict__ img_ln_b,
                const _Float16* __restrict__ img_wT, const float* __restrict__ img_b,
                const _Float16* __restrict__ fuse_wT, const float* __restrict__ fuse_b,
                float* __restrict__ out, int ntok) {
  __shared__ alignas(32) _Float16 aF[16 * TXT_D];   // LN'd rows (txt; reused for img)
  __shared__ alignas(32) _Float16 cat[16 * H_];     // relu(txt)|relu(img) concat, f16
  __shared__ float    red[2 * 128];
  __shared__ float    mu[16], rs[16];
  __shared__ int      ids[16];

  const int tid  = threadIdx.x;
  const int wave = tid >> 5;
  const int tok0 = blockIdx.x * 16;

  if (tid < 16) {
    int t = tok0 + tid;
    ids[tid] = (t < ntok) ? ids_all[t] : 0;
  }
  __syncthreads();

  // ---------- TXT branch: LN -> f16 -> GEMM 16x768 @ 768x64 ----------
  {
    const int row = tid >> 3, sub = tid & 7;
    const float* src = txt_table + (size_t)ids[row] * TXT_D;
    float s = 0.f, s2 = 0.f;
    for (int k = sub; k < TXT_D; k += 8) { float v = src[k]; s += v; s2 += v * v; }
    red[tid] = s; red[128 + tid] = s2;
    __syncthreads();
    if (sub == 0) {
      float a = 0.f, b = 0.f;
      for (int j = 0; j < 8; ++j) { a += red[row * 8 + j]; b += red[128 + row * 8 + j]; }
      float m = a / TXT_D;
      mu[row] = m;
      rs[row] = rsqrtf(b / TXT_D - m * m + 1e-5f);
    }
    __syncthreads();
    for (int idx = tid; idx < 16 * TXT_D; idx += 128) {
      int r = idx / TXT_D, k = idx - r * TXT_D;
      float v = txt_table[(size_t)ids[r] * TXT_D + k];
      v = (v - mu[r]) * rs[r] * txt_ln_s[k] + txt_ln_b[k];
      aF[idx] = (_Float16)v;
    }
    __syncthreads();

    v8f c = {};
    const int n0 = wave * 16;
    for (int k0 = 0; k0 < TXT_D; k0 += 32) {
      v16h a, b;
      load_a_f16(aF, TXT_D, k0, a);
      load_b_wt(txt_wT, TXT_D, k0, n0, b);
      c = wmma_f16(a, b, c);
    }
    const int l = tid & 31, n = n0 + (l & 15), mo = (l >> 4) * 8;
    const float bv = txt_b[n];
#pragma unroll
    for (int r = 0; r < 8; ++r)
      cat[(mo + r) * H_ + n] = (_Float16)fmaxf(c[r] + bv, 0.f);
  }
  __syncthreads();

  // ---------- IMG branch: LN -> f16 -> GEMM 16x512 @ 512x64 ----------
  {
    const int row = tid >> 3, sub = tid & 7;
    const float* src = img_table + (size_t)ids[row] * IMG_D;
    float s = 0.f, s2 = 0.f;
    for (int k = sub; k < IMG_D; k += 8) { float v = src[k]; s += v; s2 += v * v; }
    red[tid] = s; red[128 + tid] = s2;
    __syncthreads();
    if (sub == 0) {
      float a = 0.f, b = 0.f;
      for (int j = 0; j < 8; ++j) { a += red[row * 8 + j]; b += red[128 + row * 8 + j]; }
      float m = a / IMG_D;
      mu[row] = m;
      rs[row] = rsqrtf(b / IMG_D - m * m + 1e-5f);
    }
    __syncthreads();
    for (int idx = tid; idx < 16 * IMG_D; idx += 128) {
      int r = idx / IMG_D, k = idx - r * IMG_D;
      float v = img_table[(size_t)ids[r] * IMG_D + k];
      v = (v - mu[r]) * rs[r] * img_ln_s[k] + img_ln_b[k];
      aF[idx] = (_Float16)v;
    }
    __syncthreads();

    v8f c = {};
    const int n0 = wave * 16;
    for (int k0 = 0; k0 < IMG_D; k0 += 32) {
      v16h a, b;
      load_a_f16(aF, IMG_D, k0, a);
      load_b_wt(img_wT, IMG_D, k0, n0, b);
      c = wmma_f16(a, b, c);
    }
    const int l = tid & 31, n = n0 + (l & 15), mo = (l >> 4) * 8;
    const float bv = img_b[n];
#pragma unroll
    for (int r = 0; r < 8; ++r)
      cat[(mo + r) * H_ + 64 + n] = (_Float16)fmaxf(c[r] + bv, 0.f);
  }
  __syncthreads();

  // ---------- FUSE: 16x128 @ 128x128, relu, write fp32 ----------
  for (int tt = 0; tt < 2; ++tt) {
    const int n0 = (wave * 2 + tt) * 16;
    v8f c = {};
    for (int k0 = 0; k0 < H_; k0 += 32) {
      v16h a, b;
      load_a_f16(cat, H_, k0, a);
      load_b_wt(fuse_wT, H_, k0, n0, b);
      c = wmma_f16(a, b, c);
    }
    const int l = tid & 31, n = n0 + (l & 15), mo = (l >> 4) * 8;
    const float bv = fuse_b[n];
#pragma unroll
    for (int r = 0; r < 8; ++r) {
      int t = tok0 + mo + r;
      if (t < ntok) out[(size_t)t * H_ + n] = fmaxf(c[r] + bv, 0.f);
    }
  }
}

// ---------------- elementwise: x = emb*sqrt(H) + pos_e, timeline mask ----------------
__global__ void scale_pos_mask_kernel(const float* __restrict__ emb,
                                      const float* __restrict__ pos_e,
                                      const int* __restrict__ seqs,
                                      float* __restrict__ X) {
  int idx = blockIdx.x * blockDim.x + threadIdx.x;
  if (idx >= BT_ * H_) return;
  int bt = idx >> 7;            // /H_
  int h  = idx & (H_ - 1);
  int t  = bt % T_;
  float v = emb[idx] * 11.3137085f + pos_e[t * H_ + h];
  X[idx] = (seqs[bt] != 0) ? v : 0.f;
}

__global__ void mask_kernel(float* __restrict__ X, const int* __restrict__ seqs) {
  int idx = blockIdx.x * blockDim.x + threadIdx.x;
  if (idx >= BT_ * H_) return;
  if (seqs[idx >> 7] == 0) X[idx] = 0.f;
}

// ---------------- layernorm over H_=128, wave per row ----------------
__global__ void __launch_bounds__(256)
ln_kernel(const float* __restrict__ in, const float* __restrict__ s,
          const float* __restrict__ b, float eps, float* __restrict__ out, int M) {
  int row = blockIdx.x * 8 + (threadIdx.x >> 5);
  if (row >= M) return;
  int lane = threadIdx.x & 31;
  const float* x = in + (size_t)row * H_;
  float v[4], sum = 0.f;
#pragma unroll
  for (int i = 0; i < 4; ++i) { v[i] = x[lane + 32 * i]; sum += v[i]; }
#pragma unroll
  for (int off = 16; off; off >>= 1) sum += __shfl_xor(sum, off, 32);
  float m = sum * (1.f / H_);
  float var = 0.f;
#pragma unroll
  for (int i = 0; i < 4; ++i) { float d = v[i] - m; var += d * d; }
#pragma unroll
  for (int off = 16; off; off >>= 1) var += __shfl_xor(var, off, 32);
  float r = rsqrtf(var * (1.f / H_) + eps);
  float* o = out + (size_t)row * H_;
#pragma unroll
  for (int i = 0; i < 4; ++i) {
    int c = lane + 32 * i;
    o[c] = (v[i] - m) * r * s[c] + b[c];
  }
}

// ---------------- specialized WMMA GEMM, K=N=128 (all transformer projections) --------
// Block: 256 threads (8 waves), computes rows [m0, m0+64) x all 128 cols.
// Both panels staged with CDNA5 async LDS-copy (global_load_async_to_lds_b128,
// ASYNCcnt): A = 64x128 fp32 (contiguous 32KB), WT = 128x128 f16 pre-transposed
// (contiguous 32KB). Inner loop is pure LDS b128 loads + v_wmma.
__global__ void __launch_bounds__(256)
gemm128_kernel(const float* __restrict__ A, const _Float16* __restrict__ WT,
               const float* __restrict__ bias, const float* __restrict__ resid,
               float* __restrict__ out, int relu) {
  __shared__ alignas(32) float    aS[64 * H_];   // 32KB fp32 activations
  __shared__ alignas(32) _Float16 wS[H_ * H_];   // 32KB f16 weights, [n][k]

  const int tid = threadIdx.x;
  const int m0  = blockIdx.x * 64;

  // --- async-stage both panels into LDS (8 x b128 each per thread)
  {
    const float* srcA = A + (size_t)m0 * H_;             // wave-uniform base (SGPR)
    unsigned aBase = (unsigned)(uintptr_t)(void*)aS;     // low 32 bits = LDS offset
    unsigned wBase = (unsigned)(uintptr_t)(void*)wS;
#pragma unroll
    for (int i = 0; i < 8; ++i) {
      unsigned off = (unsigned)(tid * 16 + i * 4096);
      unsigned lds = aBase + off;
      asm volatile("global_load_async_to_lds_b128 %0, %1, %2"
                   :: "v"(lds), "v"(off), "s"(srcA) : "memory");
    }
#pragma unroll
    for (int i = 0; i < 8; ++i) {
      unsigned off = (unsigned)(tid * 16 + i * 4096);
      unsigned lds = wBase + off;
      asm volatile("global_load_async_to_lds_b128 %0, %1, %2"
                   :: "v"(lds), "v"(off), "s"(WT) : "memory");
    }
  }
  asm volatile("s_wait_asynccnt 0" ::: "memory");
  __syncthreads();

  const int l     = tid & 31;
  const int wave  = tid >> 5;
  const int mt    = wave >> 1;          // 0..3   (16-row tile within panel)
  const int nbase = (wave & 1) * 4;     // n-tile group: 0..3 or 4..7
  const int kbA   = (l >> 4) * 16;
  const int kbB   = (l >> 4) * 16;

  v8f acc[4] = {};
  for (int k0 = 0; k0 < H_; k0 += 32) {
    // A fragment from LDS fp32: two aligned float4x2 runs per lane
    v16h a;
    {
      const float* base = aS + (mt * 16 + (l & 15)) * H_ + k0 + kbA;
      float4 p0 = *(const float4*)(base);
      float4 p1 = *(const float4*)(base + 4);
      float4 p2 = *(const float4*)(base + 8);
      float4 p3 = *(const float4*)(base + 12);
      a[0]=(_Float16)p0.x;  a[1]=(_Float16)p0.y;  a[2]=(_Float16)p0.z;  a[3]=(_Float16)p0.w;
      a[4]=(_Float16)p1.x;  a[5]=(_Float16)p1.y;  a[6]=(_Float16)p1.z;  a[7]=(_Float16)p1.w;
      a[8]=(_Float16)p2.x;  a[9]=(_Float16)p2.y;  a[10]=(_Float16)p2.z; a[11]=(_Float16)p2.w;
      a[12]=(_Float16)p3.x; a[13]=(_Float16)p3.y; a[14]=(_Float16)p3.z; a[15]=(_Float16)p3.w;
    }
#pragma unroll
    for (int q = 0; q < 4; ++q) {
      const int n = (nbase + q) * 16 + (l & 15);
      v16h b = *(const v16h*)(wS + n * H_ + k0 + kbB);   // one 32B LDS load
      acc[q] = wmma_f16(a, b, acc[q]);
    }
  }

  // --- epilogue: bias / relu / residual
  const int mo = (l >> 4) * 8;
#pragma unroll
  for (int q = 0; q < 4; ++q) {
    const int n  = (nbase + q) * 16 + (l & 15);
    const float bv = bias ? bias[n] : 0.f;
#pragma unroll
    for (int r = 0; r < 8; ++r) {
      int m = m0 + mt * 16 + mo + r;
      float v = acc[q][r] + bv;
      if (relu) v = fmaxf(v, 0.f);
      if (resid) v += resid[(size_t)m * H_ + n];
      out[(size_t)m * H_ + n] = v;
    }
  }
}

// ---------------- attention: scores = (q @ k^T)/8 with causal -1e9 ----------------
__global__ void __launch_bounds__(256)
attn_scores_kernel(const float* __restrict__ q, const float* __restrict__ k,
                   float* __restrict__ S) {
  const int wave = threadIdx.x >> 5;
  const int tile = blockIdx.x * 8 + wave;
  const int total = B_ * NH_ * TT_ * TT_;
  if (tile >= total) return;
  const int bh = tile / (TT_ * TT_);
  const int r  = tile - bh * (TT_ * TT_);
  const int t0 = (r / TT_) * 16, s0 = (r % TT_) * 16;
  const int b = bh >> 1, h = bh & 1;
  const float* q0 = q + (size_t)b * T_ * H_ + h * DH_;
  const float* k0p = k + (size_t)b * T_ * H_ + h * DH_;
  v8f c = {};
  for (int kk = 0; kk < DH_; kk += 32) {
    v16h a, bb;
    load_a_glb(q0, H_, t0, kk, DH_, T_, a);
    load_bt_glb(k0p, H_, kk, s0, T_, bb);
    c = wmma_f16(a, bb, c);
  }
  const int l = threadIdx.x & 31, n = s0 + (l & 15), mo = (l >> 4) * 8;
  float* Sr = S + (size_t)bh * T_ * T_;
#pragma unroll
  for (int rr = 0; rr < 8; ++rr) {
    int t = t0 + mo + rr;
    if (t < T_ && n < T_)
      Sr[(size_t)t * T_ + n] = (n <= t) ? c[rr] * 0.125f : -1e9f;
  }
}

// ---------------- softmax over last dim (T_=200), wave per row ----------------
__global__ void __launch_bounds__(256)
softmax_kernel(float* __restrict__ S, int rows) {
  int row = blockIdx.x * 8 + (threadIdx.x >> 5);
  if (row >= rows) return;
  int lane = threadIdx.x & 31;
  float* x = S + (size_t)row * T_;
  float v[7], mx = -3.4e38f;
#pragma unroll
  for (int i = 0; i < 7; ++i) {
    int s = lane + 32 * i;
    v[i] = (s < T_) ? x[s] : -3.4e38f;
    mx = fmaxf(mx, v[i]);
  }
#pragma unroll
  for (int off = 16; off; off >>= 1) mx = fmaxf(mx, __shfl_xor(mx, off, 32));
  float sum = 0.f;
#pragma unroll
  for (int i = 0; i < 7; ++i) {
    v[i] = __expf(v[i] - mx);
    if (lane + 32 * i < T_) sum += v[i];
  }
#pragma unroll
  for (int off = 16; off; off >>= 1) sum += __shfl_xor(sum, off, 32);
  float inv = 1.f / sum;
#pragma unroll
  for (int i = 0; i < 7; ++i) {
    int s = lane + 32 * i;
    if (s < T_) x[s] = v[i] * inv;
  }
}

// ---------------- O = attn @ v (per b,h), merged-head layout ----------------
__global__ void __launch_bounds__(256)
attn_av_kernel(const float* __restrict__ S, const float* __restrict__ v,
               float* __restrict__ O) {
  const int NTI = DH_ / 16;  // 4
  const int wave = threadIdx.x >> 5;
  const int tile = blockIdx.x * 8 + wave;
  const int total = B_ * NH_ * TT_ * NTI;
  if (tile >= total) return;
  const int bh = tile / (TT_ * NTI);
  const int r  = tile - bh * (TT_ * NTI);
  const int t0 = (r / NTI) * 16, n0 = (r % NTI) * 16;
  const int b = bh >> 1, h = bh & 1;
  const float* A0 = S + (size_t)bh * T_ * T_;
  const float* V0 = v + (size_t)b * T_ * H_ + h * DH_;
  v8f c = {};
  for (int k0 = 0; k0 < 224; k0 += 32) {   // K=200 padded to 224, bounds-masked
    v16h a, bb;
    load_a_glb(A0, T_, t0, k0, T_, T_, a);
    load_b_glb(V0, H_, k0, n0, T_, DH_, bb);
    c = wmma_f16(a, bb, c);
  }
  const int l = threadIdx.x & 31, n = n0 + (l & 15), mo = (l >> 4) * 8;
#pragma unroll
  for (int rr = 0; rr < 8; ++rr) {
    int t = t0 + mo + rr;
    if (t < T_) O[(size_t)(b * T_ + t) * H_ + h * DH_ + n] = c[rr];
  }
}

// ---------------- logits: wave per token, dual dot over H ----------------
__global__ void __launch_bounds__(256)
logits_kernel(const float* __restrict__ feats, const float* __restrict__ ep,
              const float* __restrict__ en, float* __restrict__ out) {
  int tok = blockIdx.x * 8 + (threadIdx.x >> 5);
  if (tok >= BT_) return;
  int lane = threadIdx.x & 31;
  const float* f = feats + (size_t)tok * H_;
  const float* p = ep + (size_t)tok * H_;
  const float* q = en + (size_t)tok * H_;
  float sp = 0.f, sn = 0.f;
#pragma unroll
  for (int i = 0; i < 4; ++i) {
    int c = lane + 32 * i;
    float fv = f[c];
    sp += fv * p[c];
    sn += fv * q[c];
  }
#pragma unroll
  for (int off = 16; off; off >>= 1) {
    sp += __shfl_xor(sp, off, 32);
    sn += __shfl_xor(sn, off, 32);
  }
  if (lane == 0) { out[tok] = sp; out[BT_ + tok] = sn; }
}

// ---------------- host orchestration ----------------
extern "C" void kernel_launch(void* const* d_in, const int* in_sizes, int n_in,
                              void* d_out, int out_size, void* d_ws, size_t ws_size,
                              hipStream_t stream) {
  const int*   log_seqs  = (const int*)d_in[0];
  const int*   pos_seqs  = (const int*)d_in[1];
  const int*   neg_seqs  = (const int*)d_in[2];
  const float* txt_table = (const float*)d_in[3];
  const float* img_table = (const float*)d_in[4];
  const float* txt_ln_s  = (const float*)d_in[5];
  const float* txt_ln_b  = (const float*)d_in[6];
  const float* txt_w     = (const float*)d_in[7];
  const float* txt_b     = (const float*)d_in[8];
  const float* img_ln_s  = (const float*)d_in[9];
  const float* img_ln_b  = (const float*)d_in[10];
  const float* img_w     = (const float*)d_in[11];
  const float* img_b     = (const float*)d_in[12];
  const float* fuse_w    = (const float*)d_in[13];
  const float* fuse_b    = (const float*)d_in[14];
  const float* pos_e     = (const float*)d_in[15];
  const float* attn_ln_s = (const float*)d_in[16];
  const float* attn_ln_b = (const float*)d_in[17];
  const float* w_q       = (const float*)d_in[18];
  const float* b_q       = (const float*)d_in[19];
  const float* w_k       = (const float*)d_in[20];
  const float* b_k       = (const float*)d_in[21];
  const float* w_v       = (const float*)d_in[22];
  const float* b_v       = (const float*)d_in[23];
  const float* w_o       = (const float*)d_in[24];
  const float* b_o       = (const float*)d_in[25];
  const float* ffn_ln_s  = (const float*)d_in[26];
  const float* ffn_ln_b  = (const float*)d_in[27];
  const float* w1        = (const float*)d_in[28];
  const float* b1        = (const float*)d_in[29];
  const float* w2        = (const float*)d_in[30];
  const float* b2        = (const float*)d_in[31];
  const float* last_ln_s = (const float*)d_in[32];
  const float* last_ln_b = (const float*)d_in[33];
  (void)in_sizes; (void)n_in; (void)out_size; (void)ws_size;

  float* ws = (float*)d_ws;
  const size_t BTH  = (size_t)BT_ * H_;
  const size_t SZ_S = (size_t)B_ * NH_ * T_ * T_;
  float* EMB0 = ws;             // item emb (log), freed after scale
  float* X    = ws + 1 * BTH;   // running activation
  float* QB   = ws + 2 * BTH;   // LN(x) / Y
  float* QP   = ws + 3 * BTH;   // q proj   (later: emb_pos)
  float* KP   = ws + 4 * BTH;   // k proj   (later: emb_neg)
  float* VP   = ws + 5 * BTH;   // v proj
  float* OB   = ws + 6 * BTH;   // attn out / FFN hidden / feats
  float* S    = ws + 7 * BTH;   // B*NH*T*T scores

  // f16 transposed weights [N][K] after S
  _Float16* wh      = (_Float16*)(ws + 7 * BTH + SZ_S);
  _Float16* txt_wT  = wh;
  _Float16* img_wT  = txt_wT + 64 * TXT_D;
  _Float16* fuse_wT = img_wT + 64 * IMG_D;
  _Float16* layT    = fuse_wT + H_ * H_;   // per layer: 6 matrices of H*H

  const int embBlocks = BT_ / 16;                 // 1600
  const int ewBlocks  = (BT_ * H_ + 255) / 256;   // elementwise
  const int lnBlocks  = (BT_ + 7) / 8;            // wave-per-row LN
  const int gemmBlk   = BT_ / 64;                 // 400 (64-row panels)
  const int scBlocks  = (B_ * NH_ * TT_ * TT_ + 7) / 8;
  const int smBlocks  = (B_ * NH_ * T_ + 7) / 8;
  const int avBlocks  = (B_ * NH_ * TT_ * (DH_ / 16) + 7) / 8;
  const int cvtBlk64  = (TXT_D * 64 + 255) / 256;
  const int cvtBlk64i = (IMG_D * 64 + 255) / 256;
  const int cvtBlkH   = (H_ * H_ + 255) / 256;

  // 0) one-time per call: transpose+convert all weights to f16 [N][K]
  wcvtT_kernel<<<cvtBlk64, 256, 0, stream>>>(txt_w, txt_wT, TXT_D, 64);
  wcvtT_kernel<<<cvtBlk64i, 256, 0, stream>>>(img_w, img_wT, IMG_D, 64);
  wcvtT_kernel<<<cvtBlkH, 256, 0, stream>>>(fuse_w, fuse_wT, H_, H_);
  for (int i = 0; i < L_; ++i) {
    _Float16* Lw = layT + (size_t)i * 6 * H_ * H_;
    wcvtT_kernel<<<cvtBlkH, 256, 0, stream>>>(w_q + (size_t)i * H_ * H_, Lw + 0 * H_ * H_, H_, H_);
    wcvtT_kernel<<<cvtBlkH, 256, 0, stream>>>(w_k + (size_t)i * H_ * H_, Lw + 1 * H_ * H_, H_, H_);
    wcvtT_kernel<<<cvtBlkH, 256, 0, stream>>>(w_v + (size_t)i * H_ * H_, Lw + 2 * H_ * H_, H_, H_);
    wcvtT_kernel<<<cvtBlkH, 256, 0, stream>>>(w_o + (size_t)i * H_ * H_, Lw + 3 * H_ * H_, H_, H_);
    wcvtT_kernel<<<cvtBlkH, 256, 0, stream>>>(w1  + (size_t)i * H_ * H_, Lw + 4 * H_ * H_, H_, H_);
    wcvtT_kernel<<<cvtBlkH, 256, 0, stream>>>(w2  + (size_t)i * H_ * H_, Lw + 5 * H_ * H_, H_, H_);
  }

  // 1) log embeddings -> EMB0, then x = emb*sqrt(H) + pos_e, masked
  item_emb_kernel<<<embBlocks, 128, 0, stream>>>(
      log_seqs, txt_table, img_table, txt_ln_s, txt_ln_b, txt_wT, txt_b,
      img_ln_s, img_ln_b, img_wT, img_b, fuse_wT, fuse_b, EMB0, BT_);
  scale_pos_mask_kernel<<<ewBlocks, 256, 0, stream>>>(EMB0, pos_e, log_seqs, X);

  // 2) transformer layers
  for (int i = 0; i < L_; ++i) {
    _Float16* Lw = layT + (size_t)i * 6 * H_ * H_;
    const _Float16* wqT = Lw + 0 * H_ * H_;
    const _Float16* wkT = Lw + 1 * H_ * H_;
    const _Float16* wvT = Lw + 2 * H_ * H_;
    const _Float16* woT = Lw + 3 * H_ * H_;
    const _Float16* w1T = Lw + 4 * H_ * H_;
    const _Float16* w2T = Lw + 5 * H_ * H_;

    ln_kernel<<<lnBlocks, 256, 0, stream>>>(X, attn_ln_s + i * H_, attn_ln_b + i * H_,
                                            1e-8f, QB, BT_);
    gemm128_kernel<<<gemmBlk, 256, 0, stream>>>(QB, wqT, b_q + i * H_, nullptr, QP, 0);
    gemm128_kernel<<<gemmBlk, 256, 0, stream>>>(X,  wkT, b_k + i * H_, nullptr, KP, 0);
    gemm128_kernel<<<gemmBlk, 256, 0, stream>>>(X,  wvT, b_v + i * H_, nullptr, VP, 0);
    attn_scores_kernel<<<scBlocks, 256, 0, stream>>>(QP, KP, S);
    softmax_kernel<<<smBlocks, 256, 0, stream>>>(S, B_ * NH_ * T_);
    attn_av_kernel<<<avBlocks, 256, 0, stream>>>(S, VP, OB);
    // x = Q + (o @ w_o + b_o)
    gemm128_kernel<<<gemmBlk, 256, 0, stream>>>(OB, woT, b_o + i * H_, QB, X, 0);
    // y = LN(x); x = y + relu(y@w1+b1)@w2 + b2; mask
    ln_kernel<<<lnBlocks, 256, 0, stream>>>(X, ffn_ln_s + i * H_, ffn_ln_b + i * H_,
                                            1e-8f, QB, BT_);
    gemm128_kernel<<<gemmBlk, 256, 0, stream>>>(QB, w1T, b1 + i * H_, nullptr, OB, 1);
    gemm128_kernel<<<gemmBlk, 256, 0, stream>>>(OB, w2T, b2 + i * H_, QB, X, 0);
    mask_kernel<<<ewBlocks, 256, 0, stream>>>(X, log_seqs);
  }

  // 3) feats = LN(x); pos/neg embeddings; logits
  ln_kernel<<<lnBlocks, 256, 0, stream>>>(X, last_ln_s, last_ln_b, 1e-8f, OB, BT_);
  item_emb_kernel<<<embBlocks, 128, 0, stream>>>(
      pos_seqs, txt_table, img_table, txt_ln_s, txt_ln_b, txt_wT, txt_b,
      img_ln_s, img_ln_b, img_wT, img_b, fuse_wT, fuse_b, QP, BT_);
  item_emb_kernel<<<embBlocks, 128, 0, stream>>>(
      neg_seqs, txt_table, img_table, txt_ln_s, txt_ln_b, txt_wT, txt_b,
      img_ln_s, img_ln_b, img_wT, img_b, fuse_wT, fuse_b, KP, BT_);
  logits_kernel<<<(BT_ + 7) / 8, 256, 0, stream>>>(OB, QP, KP, (float*)d_out);
}
